// FastSpatialWindowAttention_Onnx_64647847740264
// MI455X (gfx1250) — matrix operand order repaired and checked by reading, not compiled, not run on hardware
//
#include <hip/hip_runtime.h>
#include <hip/hip_bf16.h>

// ---------- types ----------
typedef __bf16 bf16_t;
typedef bf16_t v16bf __attribute__((ext_vector_type(16)));
typedef bf16_t v8bf  __attribute__((ext_vector_type(8)));
typedef bf16_t v4bf  __attribute__((ext_vector_type(4)));
typedef float  v8f   __attribute__((ext_vector_type(8)));

union FragU { v16bf v; v8bf h[2]; };

__device__ __forceinline__ v8f wmma_bf16(const FragU& a, const FragU& b, v8f c) {
    // D = A(16x32 bf16) * B(32x16 bf16) + C(16x16 f32)
    return __builtin_amdgcn_wmma_f32_16x16x32_bf16(
        false, a.v, false, b.v, (short)0, c, false, false);
}

// async copy 16B global -> LDS (gfx1250 GLOBAL_LOAD_ASYNC_TO_LDS_B128, GV mode)
__device__ __forceinline__ void async_g2l_b128(uint32_t lds_off, const void* gaddr) {
    asm volatile("global_load_async_to_lds_b128 %0, %1, off"
                 :: "v"(lds_off), "v"(gaddr) : "memory");
}
__device__ __forceinline__ void wait_async0() {
    asm volatile("s_wait_asynccnt 0" ::: "memory");
}

// ---------- problem constants ----------
#define BB   8
#define HH   250
#define WW   250
#define LL   62500      // H*W
#define CC   256
#define NHD  8
#define HD   32
#define WS2  64
#define NW   8192       // B * 32 * 32 padded windows
#define QSCALE 0.17677669529663687f   // 32^-0.5

// ---------- workspace layout (bytes) ----------
#define OFF_XW     ((size_t)0)                       // NW*64*256 bf16 = 256 MB (reused as attn-out)
#define OFF_QKVW   ((size_t)268435456)               // 768*256 bf16
#define OFF_PROJW  (OFF_QKVW + 393216)               // 256*256 bf16
#define OFF_Q      (OFF_PROJW + 131072)              // NW*NH*64*32 bf16 = 256 MB
#define OFF_K      (OFF_Q + 268435456)
#define OFF_VT     (OFF_K + 268435456)               // [win][head][d][j]

// =====================================================================
// Kernel 1: gather 8x8 windows from x (f32), zero-pad, convert to bf16
// =====================================================================
__global__ __launch_bounds__(256) void k_gather(const float* __restrict__ x,
                                                bf16_t* __restrict__ xw) {
    int tid = threadIdx.x;
    int row = blockIdx.x * 4 + (tid >> 6);      // global window-row (win*64 + m)
    int c4  = (tid & 63) * 4;
    int win = row >> 6, m = row & 63;
    int b = win >> 10, rem = win & 1023;
    int wy = rem >> 5, wx = rem & 31;
    int iy = m >> 3, ix = m & 7;
    int y = wy * 8 + iy, xx = wx * 8 + ix;
    float4 val = make_float4(0.f, 0.f, 0.f, 0.f);
    if (y < HH && xx < WW)
        val = *(const float4*)(x + (((size_t)b * LL + (size_t)y * WW + xx) * CC + c4));
    v4bf o;
    o[0] = (bf16_t)val.x; o[1] = (bf16_t)val.y; o[2] = (bf16_t)val.z; o[3] = (bf16_t)val.w;
    *(v4bf*)(xw + (size_t)row * CC + c4) = o;
}

// =====================================================================
// Kernel 2: convert weights f32 -> bf16
// =====================================================================
__global__ __launch_bounds__(256) void k_wconv(const float* __restrict__ qkv_w,
                                               const float* __restrict__ proj_w,
                                               bf16_t* __restrict__ qkvw,
                                               bf16_t* __restrict__ projw) {
    int i = blockIdx.x * 256 + threadIdx.x;
    if (i < 768 * 256) qkvw[i] = (bf16_t)qkv_w[i];
    else               projw[i - 768 * 256] = (bf16_t)proj_w[i - 768 * 256];
}

// =====================================================================
// Kernel 3: QKV GEMM per window: (64x256)@(256x768)^T, q scaled, v transposed
// 256 threads = 8 waves; wave -> (mtile = w&3, ngroup = w>>2 of 24 ntiles)
// B fragments double-buffered across N-tiles to hide load latency.
// =====================================================================
__global__ __launch_bounds__(256, 1) void k_qkv(const bf16_t* __restrict__ xw,
                                                const bf16_t* __restrict__ qkvw,
                                                bf16_t* __restrict__ qs,
                                                bf16_t* __restrict__ ks,
                                                bf16_t* __restrict__ vT) {
    __shared__ __align__(16) bf16_t sx[WS2 * CC];   // 32 KB
    int tid = threadIdx.x, lane = tid & 31, w = tid >> 5;
    int win = blockIdx.x;

    // stage window into LDS via async DMA (2048 x 16B), bypassing VGPRs
    {
        uint32_t sbase = (uint32_t)(uintptr_t)sx;   // low 32 bits = LDS offset
        const char* g = (const char*)(xw + (size_t)win * WS2 * CC);
#pragma unroll
        for (int i = 0; i < 8; i++) {
            int e = (tid + i * 256) * 16;
            async_g2l_b128(sbase + e, g + e);
        }
        wait_async0();
    }
    __syncthreads();

    int mtile = w & 3, ngroup = w >> 2;
    int arow = mtile * 16 + (lane & 15);
    int hi = lane >> 4;

    // A fragments for this wave's mtile across K=256 (8 chunks of 32)
    FragU a[8];
#pragma unroll
    for (int kc = 0; kc < 8; kc++) {
        int e0 = kc * 32 + hi * 8;
        a[kc].h[0] = *(const v8bf*)(sx + arow * CC + e0);
        a[kc].h[1] = *(const v8bf*)(sx + arow * CC + e0 + 16);
    }

    int mBase = mtile * 16 + (hi << 3);

    auto loadB = [&](FragU (&bw)[8], int n0) {
#pragma unroll
        for (int kc = 0; kc < 8; kc++) {
            const bf16_t* wp = qkvw + (size_t)(n0 + (lane & 15)) * CC + kc * 32 + hi * 8;
            bw[kc].h[0] = *(const v8bf*)wp;
            bw[kc].h[1] = *(const v8bf*)(wp + 16);
        }
    };
    auto storeTile = [&](const v8f& acc, int n0) {
        int which = __builtin_amdgcn_readfirstlane(n0) >> 8;   // 0=q 1=k 2=v (wave-uniform)
        int head  = (n0 >> 5) & 7;
        int d     = (n0 & 31) + (lane & 15);
        size_t base = ((size_t)win * NHD + head) * (size_t)(WS2 * HD);
        if (which == 0) {
#pragma unroll
            for (int r = 0; r < 8; r++)
                qs[base + (size_t)(mBase + r) * HD + d] = (bf16_t)(acc[r] * QSCALE);
        } else if (which == 1) {
#pragma unroll
            for (int r = 0; r < 8; r++)
                ks[base + (size_t)(mBase + r) * HD + d] = (bf16_t)acc[r];
        } else {
#pragma unroll
            for (int r = 0; r < 8; r++)
                vT[base + (size_t)d * WS2 + (mBase + r)] = (bf16_t)acc[r];
        }
    };

    FragU b0[8], b1[8];
    int nbase = ngroup * 384;           // ngroup*24 tiles * 16
    loadB(b0, nbase);
    for (int nt = 0; nt < 24; nt += 2) {
        int n0 = nbase + nt * 16;
        loadB(b1, n0 + 16);             // prefetch odd tile while even computes
        v8f acc = {};
#pragma unroll
        for (int kc = 0; kc < 8; kc++) acc = wmma_bf16(a[kc], b0[kc], acc);
        storeTile(acc, n0);
        if (nt + 2 < 24) loadB(b0, n0 + 32);   // prefetch next even tile
        v8f acc2 = {};
#pragma unroll
        for (int kc = 0; kc < 8; kc++) acc2 = wmma_bf16(a[kc], b1[kc], acc2);
        storeTile(acc2, n0 + 16);
    }
}

// =====================================================================
// Kernel 4: attention per (win, head): S = qk^T + bias; softmax; P@V
// 128 threads = 4 waves, wave w owns mtile w.
// =====================================================================
__global__ __launch_bounds__(128, 1) void k_attn(const bf16_t* __restrict__ qs,
                                                 const bf16_t* __restrict__ ks,
                                                 const bf16_t* __restrict__ vT,
                                                 const float* __restrict__ rel_bias,
                                                 bf16_t* __restrict__ ao) {
    __shared__ float sbias[WS2 * WS2];              // 16 KB
    __shared__ __align__(16) bf16_t sp[WS2 * WS2];  // 8 KB
    int tid = threadIdx.x, lane = tid & 31, w = tid >> 5;
    int win = blockIdx.x >> 3, head = blockIdx.x & 7;
    int hi = lane >> 4;

    // stage rel_bias[head] (64x64 f32) into LDS
    const float4* g4 = (const float4*)(rel_bias + (size_t)head * WS2 * WS2);
    float4* s4 = (float4*)sbias;
#pragma unroll
    for (int i = 0; i < 8; i++) s4[tid + i * 128] = g4[tid + i * 128];

    size_t hb = ((size_t)win * NHD + head) * (size_t)(WS2 * HD);

    // A fragment: q rows of this wave's mtile (K = HD = 32 -> single chunk)
    const bf16_t* qp = qs + hb + (size_t)(w * 16 + (lane & 15)) * HD;
    FragU aq; aq.h[0] = *(const v8bf*)(qp + hi * 8); aq.h[1] = *(const v8bf*)(qp + hi * 8 + 16);

    // preload all 4 K-side B fragments, then 4 WMMAs
    FragU bk[4];
#pragma unroll
    for (int nt = 0; nt < 4; nt++) {
        const bf16_t* kp = ks + hb + (size_t)(nt * 16 + (lane & 15)) * HD + hi * 8;
        bk[nt].h[0] = *(const v8bf*)kp; bk[nt].h[1] = *(const v8bf*)(kp + 16);
    }
    v8f accS[4];
#pragma unroll
    for (int nt = 0; nt < 4; nt++) { v8f z = {}; accS[nt] = wmma_bf16(aq, bk[nt], z); }
    __syncthreads();

    int mBase = w * 16 + (hi << 3);
#pragma unroll
    for (int nt = 0; nt < 4; nt++) {
        int ki = nt * 16 + (lane & 15);
#pragma unroll
        for (int r = 0; r < 8; r++) accS[nt][r] += sbias[(mBase + r) * WS2 + ki];
    }

    // row softmax: row spread over 4 regs x 16 lanes (lanes 0-15 / 16-31 groups)
#pragma unroll
    for (int r = 0; r < 8; r++) {
        float mx = fmaxf(fmaxf(accS[0][r], accS[1][r]), fmaxf(accS[2][r], accS[3][r]));
#pragma unroll
        for (int msk = 1; msk < 16; msk <<= 1) mx = fmaxf(mx, __shfl_xor(mx, msk, 32));
        float sum = 0.f;
#pragma unroll
        for (int nt = 0; nt < 4; nt++) { accS[nt][r] = __expf(accS[nt][r] - mx); sum += accS[nt][r]; }
#pragma unroll
        for (int msk = 1; msk < 16; msk <<= 1) sum += __shfl_xor(sum, msk, 32);
        float inv = 1.f / sum;
#pragma unroll
        for (int nt = 0; nt < 4; nt++) accS[nt][r] *= inv;
    }

    // write P to LDS (bf16) so it can be re-read in A-fragment layout
#pragma unroll
    for (int nt = 0; nt < 4; nt++) {
        int ki = nt * 16 + (lane & 15);
#pragma unroll
        for (int r = 0; r < 8; r++) sp[(mBase + r) * WS2 + ki] = (bf16_t)accS[nt][r];
    }
    __syncthreads();

    // O = P(64x64) @ V(64x32): 2 K-chunks, 2 N-tiles
    FragU ap[2], bv[2][2];
#pragma unroll
    for (int jc = 0; jc < 2; jc++) {
        const bf16_t* pp = sp + (size_t)(w * 16 + (lane & 15)) * WS2 + jc * 32 + hi * 8;
        ap[jc].h[0] = *(const v8bf*)pp; ap[jc].h[1] = *(const v8bf*)(pp + 16);
#pragma unroll
        for (int nt = 0; nt < 2; nt++) {
            int d = nt * 16 + (lane & 15);
            const bf16_t* vp = vT + hb + (size_t)d * WS2 + jc * 32 + hi * 8;
            bv[jc][nt].h[0] = *(const v8bf*)vp; bv[jc][nt].h[1] = *(const v8bf*)(vp + 16);
        }
    }
    v8f accO[2]; accO[0] = {}; accO[1] = {};
#pragma unroll
    for (int jc = 0; jc < 2; jc++)
#pragma unroll
        for (int nt = 0; nt < 2; nt++)
            accO[nt] = wmma_bf16(ap[jc], bv[jc][nt], accO[nt]);

    size_t wb = (size_t)win * WS2 * CC;
#pragma unroll
    for (int nt = 0; nt < 2; nt++) {
        int d = nt * 16 + (lane & 15);
#pragma unroll
        for (int r = 0; r < 8; r++)
            ao[wb + (size_t)(mBase + r) * CC + head * HD + d] = (bf16_t)accO[nt][r];
    }
}

// =====================================================================
// Kernel 5: proj GEMM (64x256)@(256x256)^T + bias, scatter valid tokens
// B fragments double-buffered across the 8 N-tiles.
// =====================================================================
__global__ __launch_bounds__(256, 1) void k_proj(const bf16_t* __restrict__ ao,
                                                 const bf16_t* __restrict__ projw,
                                                 const float* __restrict__ proj_b,
                                                 float* __restrict__ out) {
    __shared__ __align__(16) bf16_t sao[WS2 * CC];  // 32 KB
    int tid = threadIdx.x, lane = tid & 31, w = tid >> 5;
    int win = blockIdx.x;
    int hi = lane >> 4;

    // async-DMA stage the attention output window into LDS
    {
        uint32_t sbase = (uint32_t)(uintptr_t)sao;
        const char* g = (const char*)(ao + (size_t)win * WS2 * CC);
#pragma unroll
        for (int i = 0; i < 8; i++) {
            int e = (tid + i * 256) * 16;
            async_g2l_b128(sbase + e, g + e);
        }
        wait_async0();
    }
    __syncthreads();

    int mtile = w & 3, ngroup = w >> 2;
    int arow = mtile * 16 + (lane & 15);
    FragU a[8];
#pragma unroll
    for (int kc = 0; kc < 8; kc++) {
        int e0 = kc * 32 + hi * 8;
        a[kc].h[0] = *(const v8bf*)(sao + arow * CC + e0);
        a[kc].h[1] = *(const v8bf*)(sao + arow * CC + e0 + 16);
    }

    int b = win >> 10, rem = win & 1023;
    int wy = rem >> 5, wx = rem & 31;
    int mBase = mtile * 16 + (hi << 3);

    auto loadB = [&](FragU (&bw)[8], int n0) {
#pragma unroll
        for (int kc = 0; kc < 8; kc++) {
            const bf16_t* wp = projw + (size_t)(n0 + (lane & 15)) * CC + kc * 32 + hi * 8;
            bw[kc].h[0] = *(const v8bf*)wp;
            bw[kc].h[1] = *(const v8bf*)(wp + 16);
        }
    };
    auto storeTile = [&](const v8f& acc, int n0) {
        int n = n0 + (lane & 15);
        float pb = proj_b[n];
#pragma unroll
        for (int r = 0; r < 8; r++) {
            int mm = mBase + r;
            int y = wy * 8 + (mm >> 3), xx = wx * 8 + (mm & 7);
            if (y < HH && xx < WW)
                out[((size_t)b * LL + (size_t)y * WW + xx) * CC + n] = acc[r] + pb;
        }
    };

    FragU b0[8], b1[8];
    int nbase = ngroup * 128;           // ngroup*8 tiles * 16
    loadB(b0, nbase);
    for (int nt = 0; nt < 8; nt += 2) {
        int n0 = nbase + nt * 16;
        loadB(b1, n0 + 16);
        v8f acc = {};
#pragma unroll
        for (int kc = 0; kc < 8; kc++) acc = wmma_bf16(a[kc], b0[kc], acc);
        storeTile(acc, n0);
        if (nt + 2 < 8) loadB(b0, n0 + 32);
        v8f acc2 = {};
#pragma unroll
        for (int kc = 0; kc < 8; kc++) acc2 = wmma_bf16(a[kc], b1[kc], acc2);
        storeTile(acc2, n0 + 16);
    }
}

// =====================================================================
extern "C" void kernel_launch(void* const* d_in, const int* in_sizes, int n_in,
                              void* d_out, int out_size, void* d_ws, size_t ws_size,
                              hipStream_t stream) {
    const float* x        = (const float*)d_in[0];
    const float* qkv_w    = (const float*)d_in[1];
    const float* proj_w   = (const float*)d_in[2];
    const float* proj_b   = (const float*)d_in[3];
    const float* rel_bias = (const float*)d_in[4];
    float* out = (float*)d_out;

    char* ws = (char*)d_ws;
    bf16_t* xw    = (bf16_t*)(ws + OFF_XW);
    bf16_t* qkvw  = (bf16_t*)(ws + OFF_QKVW);
    bf16_t* projw = (bf16_t*)(ws + OFF_PROJW);
    bf16_t* qs    = (bf16_t*)(ws + OFF_Q);
    bf16_t* ks    = (bf16_t*)(ws + OFF_K);
    bf16_t* vT    = (bf16_t*)(ws + OFF_VT);
    bf16_t* ao    = xw;   // reuse: xw fully consumed by k_qkv before k_attn writes ao

    k_gather<<<NW * WS2 / 4, 256, 0, stream>>>(x, xw);
    k_wconv<<<(768 * 256 + 256 * 256) / 256, 256, 0, stream>>>(qkv_w, proj_w, qkvw, projw);
    k_qkv<<<NW, 256, 0, stream>>>(xw, qkvw, qs, ks, vT);
    k_attn<<<NW * NHD, 128, 0, stream>>>(qs, ks, vT, rel_bias, ao);
    k_proj<<<NW, 256, 0, stream>>>(ao, projw, proj_b, out);
}
